// MultiHeadSelfAttention_31739808317965
// MI455X (gfx1250) — matrix operand order repaired and checked by reading, not compiled, not run on hardware
//
#include <hip/hip_runtime.h>

#define BB 2
#define LL 2048
#define DMODEL 1024
#define NH 16
#define DH 64

typedef __attribute__((ext_vector_type(16))) _Float16 v16h;
typedef __attribute__((ext_vector_type(8)))  _Float16 v8h;
typedef __attribute__((ext_vector_type(8)))  float    v8f;
typedef __attribute__((ext_vector_type(4)))  int      v4i;

union C8  { v8f v; float f[8]; };
union A16 { v16h v; v8h h8[2]; };

// ---- CDNA5 async global->LDS path (guarded; fallback = register copy) -----
#if defined(__has_builtin)
#  if __has_builtin(__builtin_amdgcn_global_load_async_to_lds_b128) && \
      __has_builtin(__builtin_amdgcn_s_wait_asynccnt)
#    define USE_ASYNC_LDS 1
#  endif
#endif

#ifdef USE_ASYNC_LDS
// Builtin signature (from hipcc diagnostic): (v4i addrspace(1)*, v4i addrspace(3)*, Imm, Imm)
// AS1 ptr: 64-bit, same value as generic.  AS3 ptr: 32-bit; generic LDS
// pointers carry the LDS byte offset in their low 32 bits (aperture rule).
typedef __attribute__((address_space(1))) v4i g_v4i;
typedef __attribute__((address_space(3))) v4i l_v4i;
#define TO_GLOBAL(p) ((g_v4i*)(uintptr_t)(p))
#define TO_LDS(p)    ((l_v4i*)(unsigned int)(uintptr_t)(p))
#endif

// A-matrix fragment, 16x32 f16 (ISA 7.12.2):
// lane 0-15: M=lane, halves 0..7 -> K 0..7,  halves 8..15 -> K 16..23
// lane16-31: M=lane-16, halves 0..7 -> K 8..15, halves 8..15 -> K 24..31
__device__ __forceinline__ v16h load_a_frag(const _Float16* base, int stride) {
  const int lane = threadIdx.x & 31;
  const int m  = lane & 15;
  const int kb = (lane >> 4) << 3;      // 0 or 8
  const _Float16* p = base + m * stride + kb;
  A16 u;
  u.h8[0] = *(const v8h*)(p);
  u.h8[1] = *(const v8h*)(p + 16);
  return u.v;
}

// B-matrix fragment, 32x16 f16, source stored transposed as [N][K]:
// lane 0-15: N=lane, halves 0..7 -> K 0..7, 8..15 -> K 16..23; lane16-31: N=lane-16, K 8..15, 24..31
__device__ __forceinline__ v16h load_bT_frag(const _Float16* baseT, int stride, int koff) {
  const int lane = threadIdx.x & 31;
  const int n  = lane & 15;
  const int kb = (lane >> 4) << 3;      // 0 or 8
  const _Float16* p = baseT + n * stride + koff + kb;
  A16 u;
  u.h8[0] = *(const v8h*)(p);
  u.h8[1] = *(const v8h*)(p + 16);
  return u.v;
}

// ---------------------------------------------------------------------------
// Tiled GEMM: out[M,N] = A[M,K] * W[N,K]^T + bias[N]
// block tile 128x128, BK=32, 256 threads = 8 waves.
// Each wave: 2 M-tiles x 4 N-tiles -> 8 WMMAs per K-step.
// ---------------------------------------------------------------------------
#define BM 128
#define BN 128
#define BK 32
#define LDA 40
#define LDB 40

template <typename AT, typename OT>
__global__ void __launch_bounds__(256)
gemm_bias_kernel(const AT* __restrict__ A, const float* __restrict__ W,
                 const float* __restrict__ bias, OT* __restrict__ out,
                 int M, int N, int K) {
  __shared__ __align__(32) _Float16 sA[BM * LDA];
  __shared__ __align__(32) _Float16 sB[BN * LDB];

  const int tid  = threadIdx.x;
  const int lane = tid & 31;
  const int wave = tid >> 5;
  const int n0 = blockIdx.x * BN;
  const int m0 = blockIdx.y * BM;
  const int mt0 = (wave >> 1) * 32;     // wave's first M tile row (0,32,64,96)
  const int tg  = (wave & 1) * 4;       // N tile group

  C8 acc[2][4];
#pragma unroll
  for (int t = 0; t < 4; ++t) {
    float bv = bias[n0 + (tg + t) * 16 + (lane & 15)];
#pragma unroll
    for (int r = 0; r < 8; ++r) { acc[0][t].f[r] = bv; acc[1][t].f[r] = bv; }
  }

  const int srow = tid >> 1;            // 0..127
  const int scol = (tid & 1) * 16;      // 0,16

  for (int k0 = 0; k0 < K; k0 += BK) {
    __syncthreads();
    {
      const AT* asrc = A + (size_t)(m0 + srow) * K + k0 + scol;
      _Float16* adst = sA + srow * LDA + scol;
#pragma unroll
      for (int j = 0; j < 16; ++j) adst[j] = (_Float16)(float)asrc[j];
      const float* wsrc = W + (size_t)(n0 + srow) * K + k0 + scol;
      _Float16* wdst = sB + srow * LDB + scol;
#pragma unroll
      for (int j = 0; j < 16; ++j) wdst[j] = (_Float16)wsrc[j];
    }
    __syncthreads();

    v16h a0 = load_a_frag(sA + mt0 * LDA, LDA);
    v16h a1 = load_a_frag(sA + (mt0 + 16) * LDA, LDA);
#pragma unroll
    for (int t = 0; t < 4; ++t) {
      v16h b = load_bT_frag(sB + (tg + t) * 16 * LDB, LDB, 0);
      acc[0][t].v = __builtin_amdgcn_wmma_f32_16x16x32_f16(
          false, a0, false, b, (short)0, acc[0][t].v, false, false);
      acc[1][t].v = __builtin_amdgcn_wmma_f32_16x16x32_f16(
          false, a1, false, b, (short)0, acc[1][t].v, false, false);
    }
  }

  const int hi8 = (lane >> 4) << 3;
#pragma unroll
  for (int sub = 0; sub < 2; ++sub) {
#pragma unroll
    for (int t = 0; t < 4; ++t) {
      int col = n0 + (tg + t) * 16 + (lane & 15);
#pragma unroll
      for (int r = 0; r < 8; ++r) {
        int row = m0 + mt0 + sub * 16 + hi8 + r;
        out[(size_t)row * N + col] = (OT)acc[sub][t].f[r];
      }
    }
  }
}

// ---------------------------------------------------------------------------
// RoPE + split into Q/K/V [B,H,L,DH] f16.  Q pre-scaled by 1/sqrt(DH).
// ---------------------------------------------------------------------------
__global__ void __launch_bounds__(256)
rope_split_kernel(const _Float16* __restrict__ qkv,
                  _Float16* __restrict__ Q, _Float16* __restrict__ K,
                  _Float16* __restrict__ V) {
  int t = blockIdx.x * 256 + threadIdx.x;     // ((b*16+h)*2048+l)*32+i
  int i = t & 31;
  int l = (t >> 5) & (LL - 1);
  int h = (t >> 16) & (NH - 1);
  int b = t >> 20;

  const _Float16* src = qkv + (size_t)(b * LL + l) * (3 * DMODEL);
  int base = h * DH + i;
  float q1 = (float)src[base],              q2 = (float)src[base + 32];
  float k1 = (float)src[DMODEL + base],     k2 = (float)src[DMODEL + base + 32];
  float v1 = (float)src[2 * DMODEL + base], v2 = (float)src[2 * DMODEL + base + 32];

  float freq = __expf(-(float)i * 0.2878231366242557f);  // ln(10000)/32
  float ang  = (float)l * freq;
  float sn, cs;
  __sincosf(ang, &sn, &cs);

  size_t o = ((size_t)((b * NH + h) * LL + l)) * DH + i;
  const float scale = 0.125f;  // 1/sqrt(64)
  Q[o]      = (_Float16)((q1 * cs - q2 * sn) * scale);
  Q[o + 32] = (_Float16)((q1 * sn + q2 * cs) * scale);
  K[o]      = (_Float16)(k1 * cs - k2 * sn);
  K[o + 32] = (_Float16)(k1 * sn + k2 * cs);
  V[o]      = (_Float16)v1;
  V[o + 32] = (_Float16)v2;
}

// ---------------------------------------------------------------------------
// Flash attention.  grid = (L/128, H, B), 256 threads = 8 waves.
// Wave w owns q-rows [q0+16w, q0+16w+16): fully private online softmax.
// K tile staged via async global->LDS (ASYNCcnt); V staged transposed.
// ---------------------------------------------------------------------------
#define KBLK 64
#define LDK 72

__global__ void __launch_bounds__(256)
attn_kernel(const _Float16* __restrict__ Q, const _Float16* __restrict__ Km,
            const _Float16* __restrict__ V, const int* __restrict__ mask,
            _Float16* __restrict__ ctx) {
  __shared__ __align__(32) _Float16 sK[KBLK * LDK];       // [key][d]
  __shared__ __align__(32) _Float16 sVt[DH * LDK];        // [d][key]
  __shared__ __align__(32) _Float16 sP[8 * 16 * LDK];     // per-wave P 16x64

  const int tid  = threadIdx.x;
  const int lane = tid & 31;
  const int wave = tid >> 5;
  const int b = blockIdx.z, h = blockIdx.y;
  const int bh = b * NH + h;
  const int q0 = blockIdx.x * 128;
  const int qrow0 = q0 + wave * 16;
  const int hi8 = (lane >> 4) << 3;

  const _Float16* qbase = Q + (size_t)bh * LL * DH + (size_t)qrow0 * DH;
  v16h aq0 = load_a_frag(qbase, DH);
  v16h aq1 = load_a_frag(qbase + 32, DH);

  float mrow[8], lsum[8];
  C8 o[4];
#pragma unroll
  for (int r = 0; r < 8; ++r) { mrow[r] = -1e30f; lsum[r] = 0.f; }
#pragma unroll
  for (int t = 0; t < 4; ++t)
#pragma unroll
    for (int r = 0; r < 8; ++r) o[t].f[r] = 0.f;

  const int nkb = (q0 + 128) / KBLK;     // uniform across the block
  const int krow = tid >> 2;             // 0..63
  const int kcol = (tid & 3) * 16;       // 0,16,32,48
  const _Float16* kvbase = Km + (size_t)bh * LL * DH;
  const _Float16* vvbase = V  + (size_t)bh * LL * DH;

  for (int kb = 0; kb < nkb; ++kb) {
    const int k0 = kb * KBLK;
    __syncthreads();
    {
      const _Float16* ksrc = kvbase + (size_t)(k0 + krow) * DH + kcol;
      const _Float16* vsrc = vvbase + (size_t)(k0 + krow) * DH + kcol;
      _Float16* kdst = sK + krow * LDK + kcol;
#ifdef USE_ASYNC_LDS
      // two contiguous b128 chunks per thread, pure byte copy -> async DMA
      __builtin_amdgcn_global_load_async_to_lds_b128(TO_GLOBAL(ksrc),
                                                     TO_LDS(kdst), 0, 0);
      __builtin_amdgcn_global_load_async_to_lds_b128(TO_GLOBAL(ksrc + 8),
                                                     TO_LDS(kdst + 8), 0, 0);
#else
#pragma unroll
      for (int j = 0; j < 16; ++j) kdst[j] = ksrc[j];
#endif
#pragma unroll
      for (int j = 0; j < 16; ++j)
        sVt[(kcol + j) * LDK + krow] = vsrc[j];   // transpose V
      if (kb + 1 < nkb) {                         // prefetch next tile
        __builtin_prefetch(ksrc + KBLK * DH, 0, 0);
        __builtin_prefetch(vsrc + KBLK * DH, 0, 0);
      }
#ifdef USE_ASYNC_LDS
      __builtin_amdgcn_s_wait_asynccnt(0);
#endif
    }
    __syncthreads();

    // S = Q * K^T (4 tiles of 16x16, K-dim 64)
    C8 s[4];
#pragma unroll
    for (int t = 0; t < 4; ++t)
#pragma unroll
      for (int r = 0; r < 8; ++r) s[t].f[r] = 0.f;
#pragma unroll
    for (int t = 0; t < 4; ++t) {
      v16h b0 = load_bT_frag(sK + (t * 16) * LDK, LDK, 0);
      s[t].v = __builtin_amdgcn_wmma_f32_16x16x32_f16(
          false, aq0, false, b0, (short)0, s[t].v, false, false);
      v16h b1 = load_bT_frag(sK + (t * 16) * LDK, LDK, 32);
      s[t].v = __builtin_amdgcn_wmma_f32_16x16x32_f16(
          false, aq1, false, b1, (short)0, s[t].v, false, false);
    }

    // causal + attn_mask, row max
    float lm[8];
#pragma unroll
    for (int r = 0; r < 8; ++r) lm[r] = -1e30f;
#pragma unroll
    for (int t = 0; t < 4; ++t) {
      int col = k0 + t * 16 + (lane & 15);
#pragma unroll
      for (int r = 0; r < 8; ++r) {
        int row = qrow0 + hi8 + r;
        float v = s[t].f[r];
        bool ok = (col <= row) &&
                  (mask[(size_t)b * LL * LL + (size_t)row * LL + col] != 0);
        v = ok ? v : -1e30f;
        s[t].f[r] = v;
        lm[r] = fmaxf(lm[r], v);
      }
    }
    // each 16x16 row lives entirely in one 16-lane half -> xor {1,2,4,8}
#pragma unroll
    for (int r = 0; r < 8; ++r) {
#pragma unroll
      for (int off = 1; off < 16; off <<= 1)
        lm[r] = fmaxf(lm[r], __shfl_xor(lm[r], off, 32));
    }
#pragma unroll
    for (int r = 0; r < 8; ++r) {
      float mn = fmaxf(mrow[r], lm[r]);
      float sc = __expf(mrow[r] - mn);
      mrow[r] = mn;
      float ps = 0.f;
#pragma unroll
      for (int t = 0; t < 4; ++t) {
        float p = __expf(s[t].f[r] - mn);
        s[t].f[r] = p;
        ps += p;
      }
#pragma unroll
      for (int off = 1; off < 16; off <<= 1)
        ps += __shfl_xor(ps, off, 32);
      lsum[r] = lsum[r] * sc + ps;
#pragma unroll
      for (int t = 0; t < 4; ++t) o[t].f[r] *= sc;
    }

    // redistribute P (C layout -> A layout) via per-wave LDS buffer
    _Float16* pw = sP + wave * 16 * LDK;
#pragma unroll
    for (int t = 0; t < 4; ++t)
#pragma unroll
      for (int r = 0; r < 8; ++r)
        pw[(hi8 + r) * LDK + t * 16 + (lane & 15)] = (_Float16)s[t].f[r];
    __syncthreads();

    // O += P * V
#pragma unroll
    for (int kc = 0; kc < 2; ++kc) {
      v16h ap = load_a_frag(pw + 32 * kc, LDK);
#pragma unroll
      for (int t = 0; t < 4; ++t) {
        v16h bv = load_bT_frag(sVt + (t * 16) * LDK, LDK, 32 * kc);
        o[t].v = __builtin_amdgcn_wmma_f32_16x16x32_f16(
            false, ap, false, bv, (short)0, o[t].v, false, false);
      }
    }
  }

  // normalize and write ctx as [B, L, H*DH] f16
#pragma unroll
  for (int t = 0; t < 4; ++t) {
    int d = t * 16 + (lane & 15);
#pragma unroll
    for (int r = 0; r < 8; ++r) {
      int row = qrow0 + hi8 + r;
      float val = o[t].f[r] / lsum[r];
      ctx[((size_t)(b * LL + row)) * DMODEL + h * DH + d] = (_Float16)val;
    }
  }
}

// ---------------------------------------------------------------------------
extern "C" void kernel_launch(void* const* d_in, const int* in_sizes, int n_in,
                              void* d_out, int out_size, void* d_ws, size_t ws_size,
                              hipStream_t stream) {
  (void)in_sizes; (void)n_in; (void)out_size; (void)ws_size;
  const float* x     = (const float*)d_in[0];
  const float* qkv_w = (const float*)d_in[1];
  const float* qkv_b = (const float*)d_in[2];
  const float* o_w   = (const float*)d_in[3];
  const float* o_b   = (const float*)d_in[4];
  const int*   amask = (const int*)d_in[5];

  const size_t nBHLd = (size_t)BB * NH * LL * DH;       // 4M halves each
  _Float16* Qh  = (_Float16*)d_ws;
  _Float16* Kh  = Qh + nBHLd;
  _Float16* Vh  = Kh + nBHLd;
  _Float16* qkv = Vh + nBHLd;                           // [4096,3072] f16
  _Float16* ctx = qkv;                                  // reuse (qkv dead after rope)

  const int Mrows = BB * LL;                            // 4096
  dim3 blk(256);

  // 1) qkv = x @ qkv_w^T + qkv_b    (f32 in, f16 out)
  gemm_bias_kernel<float, _Float16>
      <<<dim3((3 * DMODEL) / BN, Mrows / BM), blk, 0, stream>>>(
          x, qkv_w, qkv_b, qkv, Mrows, 3 * DMODEL, DMODEL);

  // 2) RoPE + head split (Q pre-scaled by 1/sqrt(Dh))
  rope_split_kernel<<<dim3((BB * NH * LL * 32) / 256), blk, 0, stream>>>(
      qkv, Qh, Kh, Vh);

  // 3) causal flash attention
  attn_kernel<<<dim3(LL / 128, NH, BB), blk, 0, stream>>>(Qh, Kh, Vh, amask, ctx);

  // 4) out = ctx @ o_w^T + o_b      (f16 in, f32 out)
  gemm_bias_kernel<_Float16, float>
      <<<dim3(DMODEL / BN, Mrows / BM), blk, 0, stream>>>(
          ctx, o_w, o_b, (float*)d_out, Mrows, DMODEL, DMODEL);
}
